// DiagonalLSTM_6545530159353
// MI455X (gfx1250) — matrix operand order repaired and checked by reading, not compiled, not run on hardware
//
#include <hip/hip_runtime.h>

typedef _Float16 v16h __attribute__((ext_vector_type(16)));
typedef float    v8f  __attribute__((ext_vector_type(8)));

#define BATCH 16
#define CIN   32
#define HDIM  128
#define WDIM  128
#define BO    32
#define GO    128      // 4*BO gate outputs
#define KTOT  96       // 32 (h shifted) + 32 (h) + 32 (x proj)
#define BW_STRIDE 100  // padded k-stride for weight LDS (bank-conflict free)
#define H_STRIDE  40   // padded c-stride for state/x LDS (bank-conflict free)

// ---- WMMA fragment loaders (wave32; contiguous k order, same in A and B) ----

// A: 16x32 f16, row-major in LDS. Lane r (&15) = row M; lanes<16 hold K 0-15,
// lanes>=16 hold K 16-31, 2 halves per VGPR.
__device__ __forceinline__ v16h load_A_frag(const _Float16* base, int stride) {
  const int lane = threadIdx.x & 31;
  const int r    = lane & 15;
  const int hi   = lane >> 4;
  const _Float16* p = base + r * stride;
  v16h a;
#pragma unroll
  for (int i = 0; i < 16; ++i) {
    const int k = hi * 16 + i;
    a[i] = p[k];
  }
  return a;
}

// B: 32x16 f16. Stored transposed in LDS: element (k,n) at baseT[n*stride+k].
// Lane n (&15) = column N; lanes<16 hold K 0-15, lanes>=16 hold K 16-31,
// consecutive K pairs per VGPR -> b32-combinable LDS reads.
__device__ __forceinline__ v16h load_B_frag(const _Float16* baseT, int stride) {
  const int lane = threadIdx.x & 31;
  const int n  = lane & 15;
  const int kb = (lane >> 4) * 16;
  const _Float16* p = baseT + n * stride + kb;
  v16h b;
#pragma unroll
  for (int i = 0; i < 16; ++i) b[i] = p[i];
  return b;
}

// ---- Phase 0: transpose+convert x (b,c,h,w) f32 -> (b,h,w,c) f16 ----
__global__ void xpose_f16(const float* __restrict__ x, _Float16* __restrict__ xt) {
  __shared__ float tile[32][33];
  const int wb = blockIdx.x;   // w block (4)
  const int h  = blockIdx.y;   // 128
  const int b  = blockIdx.z;   // 16
  const int tx = threadIdx.x;  // 0..31
  const int ty = threadIdx.y;  // 0..7
  const int w0 = wb * 32;
#pragma unroll
  for (int cy = 0; cy < 4; ++cy) {
    const int c = cy * 8 + ty;
    tile[c][tx] = x[(((size_t)b * CIN + c) * HDIM + h) * WDIM + w0 + tx]; // coalesced over w
  }
  __syncthreads();
#pragma unroll
  for (int wy = 0; wy < 4; ++wy) {
    const int w = w0 + wy * 8 + ty;
    xt[(((size_t)b * HDIM + h) * WDIM + w) * CIN + tx] = (_Float16)tile[tx][wy * 8 + ty]; // coalesced over c
  }
}

// ---- Phase 1: persistent wavefront LSTM, one workgroup per batch ----
// __launch_bounds__(256, 4): all 8 waves must be co-resident (per-step
// barriers); cap VGPRs at <=256/wave so the workgroup fits in CU mode.
__global__ void __launch_bounds__(256, 4) diag_lstm(
    const _Float16* __restrict__ xt,
    const float* __restrict__ W2g, const float* __restrict__ b2g,
    const float* __restrict__ W1g, const float* __restrict__ b1g,
    float* __restrict__ out) {
  __shared__ _Float16 BWt[GO * BW_STRIDE];                 // [o][k] k-contiguous: W1_prev|W1_cur|W2^T
  __shared__ float    bias_s[GO];                          // b1+b2
  __shared__ _Float16 Hbuf[2][(HDIM + 1) * H_STRIDE];      // double-buffered h state, row 0 = zeros (h=-1)
  __shared__ _Float16 Xd[HDIM * H_STRIDE];                 // diagonal x slice for current t

  const int b      = blockIdx.x;
  const int tid    = threadIdx.x;
  const int lane   = tid & 31;
  const int wv     = tid >> 5;     // 8 waves, wave wv owns rows 16wv..16wv+15
  const int n15    = lane & 15;
  const int hiHalf = lane >> 4;

  // Stage weights/bias into LDS (f16), zero both state buffers.
  for (int idx = tid; idx < GO * KTOT; idx += 256) {
    const int o = idx / KTOT, k = idx % KTOT;
    float v;
    if (k < 32)      v = W1g[(o * 32 + k) * 2 + 0];        // W1_prev
    else if (k < 64) v = W1g[(o * 32 + (k - 32)) * 2 + 1]; // W1_cur
    else             v = W2g[o * 32 + (k - 64)];           // W2^T
    BWt[o * BW_STRIDE + k] = (_Float16)v;
  }
  for (int idx = tid; idx < GO; idx += 256) bias_s[idx] = b1g[idx] + b2g[idx];
  for (int idx = tid; idx < 2 * (HDIM + 1) * H_STRIDE; idx += 256)
    (&Hbuf[0][0])[idx] = (_Float16)0.f;
  __syncthreads();

  float biasv[8];
#pragma unroll
  for (int tn = 0; tn < 8; ++tn) biasv[tn] = bias_s[n15 + 16 * tn];

  float cst[16];                                           // cell state, registers, persistent over t
#pragma unroll
  for (int i = 0; i < 16; ++i) cst[i] = 0.f;

  const int xrow  = tid >> 1;   // row loaded by this thread
  const int xpart = tid & 1;    // half of the 32-channel vector

#pragma unroll 1
  for (int t = 0; t < WDIM; ++t) {
    // Stage diagonal x slice: Xd[h][c] = x16[b][h][t-h][c], zero when t<h.
    {
      const int wcol = t - xrow;
      float4* dst = (float4*)&Xd[xrow * H_STRIDE + xpart * 16];
      if (wcol >= 0) {
        const float4* src = (const float4*)(xt +
            (((size_t)b * HDIM + xrow) * WDIM + wcol) * CIN + xpart * 16);
        dst[0] = src[0];
        dst[1] = src[1];
        if (wcol + 1 < WDIM) __builtin_prefetch(src + 4, 0, 0); // next t -> global_prefetch
      } else {
        const float4 z = make_float4(0.f, 0.f, 0.f, 0.f);
        dst[0] = z; dst[1] = z;
      }
    }
    __syncthreads();  // Xd ready; prior step's H writes visible

    const int rb = t & 1;
    const _Float16* Hr = &Hbuf[rb][0];
    const _Float16* A0 = Hr + (16 * wv) * H_STRIDE;        // state rows h-1 (zero row handles h=0)
    const _Float16* A1 = Hr + (16 * wv + 1) * H_STRIDE;    // state rows h
    const _Float16* A2 = &Xd[(16 * wv) * H_STRIDE];        // x diagonal rows h

    const v16h a0 = load_A_frag(A0, H_STRIDE);
    const v16h a1 = load_A_frag(A1, H_STRIDE);
    const v16h a2 = load_A_frag(A2, H_STRIDE);

    _Float16* Hw = &Hbuf[rb ^ 1][0];

    // Two column groups (bo 0-15, 16-31): only the 4 gate tiles for one
    // group are live at a time -> ~32 live acc VGPRs instead of 64.
#pragma unroll
    for (int tn = 0; tn < 2; ++tn) {
      v8f acc[4];                                          // o, f, i, g tiles
#pragma unroll
      for (int g = 0; g < 4; ++g) {
        const int tile = tn + 2 * g;                       // N tile: tn, tn+2, tn+4, tn+6
        v8f c;
#pragma unroll
        for (int i = 0; i < 8; ++i) c[i] = biasv[tile];    // bias folded into C
        const _Float16* bt = &BWt[(16 * tile) * BW_STRIDE];
        const v16h bf0 = load_B_frag(bt,      BW_STRIDE);  // W1_prev
        const v16h bf1 = load_B_frag(bt + 32, BW_STRIDE);  // W1_cur
        const v16h bf2 = load_B_frag(bt + 64, BW_STRIDE);  // W2^T
        c = __builtin_amdgcn_wmma_f32_16x16x32_f16(false, a0, false, bf0, (short)0, c, false, false);
        c = __builtin_amdgcn_wmma_f32_16x16x32_f16(false, a1, false, bf1, (short)0, c, false, false);
        c = __builtin_amdgcn_wmma_f32_16x16x32_f16(false, a2, false, bf2, (short)0, c, false, false);
        acc[g] = c;
      }

      // Gate math: lane holds o/f/i/g for the same (h,bo) across acc[0..3].
#pragma unroll
      for (int r = 0; r < 8; ++r) {
        const float go = acc[0][r];
        const float gf = acc[1][r];
        const float gi = acc[2][r];
        const float gg = acc[3][r];
        const float so = 1.f / (1.f + __expf(-go));
        const float sf = 1.f / (1.f + __expf(-gf));
        const float si = 1.f / (1.f + __expf(-gi));
        const float tg = 1.f - 2.f / (__expf(2.f * gg) + 1.f);
        const float cv = sf * cst[tn * 8 + r] + si * tg;
        cst[tn * 8 + r] = cv;
        const float hv = so * (1.f - 2.f / (__expf(2.f * cv) + 1.f));
        const int h  = 16 * wv + r + hiHalf * 8;           // C/D layout: M = r + 8*(lane>=16)
        const int bo = n15 + 16 * tn;                      // C/D layout: N = lane&15
        Hw[(h + 1) * H_STRIDE + bo] = (_Float16)hv;
        out[(((size_t)b * BO + bo) * HDIM + h) * WDIM + t] = hv;
      }
    }
    __syncthreads();  // protect Xd/H before next step's staging
  }
}

extern "C" void kernel_launch(void* const* d_in, const int* in_sizes, int n_in,
                              void* d_out, int out_size, void* d_ws, size_t ws_size,
                              hipStream_t stream) {
  (void)in_sizes; (void)n_in; (void)out_size; (void)ws_size;
  const float* x  = (const float*)d_in[0];
  const float* W2 = (const float*)d_in[1];
  const float* b2 = (const float*)d_in[2];
  const float* W1 = (const float*)d_in[3];
  const float* b1 = (const float*)d_in[4];
  float* out = (float*)d_out;
  _Float16* xt = (_Float16*)d_ws;  // 16*128*128*32 f16 = 16 MB

  hipLaunchKernelGGL(xpose_f16, dim3(WDIM / 32, HDIM, BATCH), dim3(32, 8), 0, stream, x, xt);
  hipLaunchKernelGGL(diag_lstm, dim3(BATCH), dim3(256), 0, stream, xt, W2, b2, W1, b1, out);
}
